// StructuralAttentionLayer_43396349558868
// MI455X (gfx1250) — matrix operand order repaired and checked by reading, not compiled, not run on hardware
//
#include <hip/hip_runtime.h>

// ---- problem constants (match reference) ----
#define B_   4
#define N_   2048
#define IND  256
#define OUTD 64
#define NH   8
#define HD   512          // NH*OUTD
#define ALPHA_ 0.2f
#define NEGV  -9e15f

typedef _Float16 v16h __attribute__((ext_vector_type(16)));
typedef _Float16 v8h  __attribute__((ext_vector_type(8)));
typedef float    v8f  __attribute__((ext_vector_type(8)));

// A-operand (16x32 f16) loader: lane l holds row m=l&15,
// elems [0..7] = K=klo..klo+7, elems[8..15] = K=klo+16..klo+23, klo=8*(l>>4)
__device__ __forceinline__ v16h load_A16x32(const _Float16* __restrict__ base,
                                            int ld, int lane, int row0, int k0) {
  int m   = lane & 15;
  int klo = (lane >> 4) << 3;
  const _Float16* p = base + (size_t)(row0 + m) * ld + k0 + klo;
  v8h lo = *(const v8h*)p;
  v8h hi = *(const v8h*)(p + 16);
  return __builtin_shufflevector(lo, hi, 0,1,2,3,4,5,6,7,8,9,10,11,12,13,14,15);
}

// B-operand (32x16 f16) loader: lane l holds col n=c0+(l&15),
// elems [0..7] = K=klo..klo+7, elems[8..15] = K=klo+16..klo+23, klo=8*(l>>4)
__device__ __forceinline__ v16h load_B32x16(const _Float16* __restrict__ base,
                                            int ld, int lane, int k0, int c0) {
  int n   = lane & 15;
  int klo = (lane >> 4) << 3;
  const _Float16* p = base + (size_t)(k0 + klo) * ld + c0 + n;
  v16h r;
#pragma unroll
  for (int s = 0; s < 8; ++s) {
    r[s]     = p[(size_t)s * ld];
    r[8 + s] = p[(size_t)(16 + s) * ld];
  }
  return r;
}

__device__ __forceinline__ v8f wmma_f16(v16h a, v16h b, v8f c) {
  return __builtin_amdgcn_wmma_f32_16x16x32_f16(false, a, false, b,
                                                (short)0, c, false, false);
}

// ---------------- stage 1: f32 -> f16 convert ----------------
__global__ void f32_to_f16_kernel(const float* __restrict__ src,
                                  _Float16* __restrict__ dst, int n) {
  int i = blockIdx.x * blockDim.x + threadIdx.x;
  if (i < n) dst[i] = (_Float16)src[i];
}

// ---------------- stage 2: projections ----------------
// grid = (512 row-tiles, 3 matrices); block = 256 (8 waves = 8 heads / col chunks)
// Q,V stored [bh][N][64]; K stored transposed [bh][64][N]
__global__ void __launch_bounds__(256)
proj_kernel(const _Float16* __restrict__ Xh,   // [8192][256]
            const _Float16* __restrict__ Wh,   // [3][256][512]
            _Float16* __restrict__ Q,
            _Float16* __restrict__ Kt,
            _Float16* __restrict__ V) {
  int lane = threadIdx.x & 31;
  int wave = threadIdx.x >> 5;          // head / 64-col chunk
  int row0 = blockIdx.x * 16;
  int mat  = blockIdx.y;
  int col0 = wave * OUTD;
  const _Float16* W = Wh + (size_t)mat * IND * HD;

  v8f acc[4] = {v8f{}, v8f{}, v8f{}, v8f{}};
#pragma unroll
  for (int k0 = 0; k0 < IND; k0 += 32) {
    v16h a = load_A16x32(Xh, IND, lane, row0, k0);
#pragma unroll
    for (int c = 0; c < 4; ++c) {
      v16h b = load_B32x16(W, HD, lane, k0, col0 + c * 16);
      acc[c] = wmma_f16(a, b, acc[c]);
    }
  }
  int b  = row0 / N_;
  int n0 = row0 % N_;
  int grp = lane >> 4, ln = lane & 15;
  int bh = b * NH + wave;
#pragma unroll
  for (int c = 0; c < 4; ++c) {
#pragma unroll
    for (int r = 0; r < 8; ++r) {
      int rowm = r + 8 * grp;
      int d    = c * 16 + ln;
      _Float16 v = (_Float16)acc[c][r];
      size_t n = (size_t)(n0 + rowm);
      if (mat == 0)      Q [((size_t)bh * N_ + n) * OUTD + d] = v;
      else if (mat == 1) Kt[((size_t)bh * OUTD + d) * N_ + n] = v;
      else               V [((size_t)bh * N_ + n) * OUTD + d] = v;
    }
  }
}

// ---------------- stage 3: flash attention ----------------
// grid = (N/128, H, B); block = 256 (8 waves x 16 query rows)
__global__ void __launch_bounds__(256)
attn_kernel(const _Float16* __restrict__ Q,
            const _Float16* __restrict__ Kt,
            const _Float16* __restrict__ V,
            const int* __restrict__ adj,
            float* __restrict__ out) {
  __shared__ _Float16 ldsP[8][16][32];   // per-wave P tile (16 rows x 32 keys)

  int lane = threadIdx.x & 31;
  int wave = threadIdx.x >> 5;
  int grp  = lane >> 4, ln = lane & 15;
  int h = blockIdx.y, b = blockIdx.z;
  int bh = b * NH + h;
  int qbase = blockIdx.x * 128 + wave * 16;

  const _Float16* Qb  = Q  + (size_t)bh * N_ * OUTD;
  const _Float16* Ktb = Kt + (size_t)bh * OUTD * N_;
  const _Float16* Vb  = V  + (size_t)bh * N_ * OUTD;
  const int* adjb = adj + (size_t)b * N_ * N_;

  // Q A-operands are loop invariant (16 rows x 64 dims -> two 16x32 operands)
  v16h aq0 = load_A16x32(Qb, OUTD, lane, qbase, 0);
  v16h aq1 = load_A16x32(Qb, OUTD, lane, qbase, 32);

  v8f acc[4] = {v8f{}, v8f{}, v8f{}, v8f{}};
  float mrow[8], lrow[8];
#pragma unroll
  for (int r = 0; r < 8; ++r) { mrow[r] = -INFINITY; lrow[r] = 0.0f; }

  for (int kb = 0; kb < N_; kb += 32) {
    // ---- S = Q K^T for 32 keys (two 16-key C tiles) ----
    v8f s0 = {}, s1 = {};
    s0 = wmma_f16(aq0, load_B32x16(Ktb, N_, lane, 0,  kb),      s0);
    s0 = wmma_f16(aq1, load_B32x16(Ktb, N_, lane, 32, kb),      s0);
    s1 = wmma_f16(aq0, load_B32x16(Ktb, N_, lane, 0,  kb + 16), s1);
    s1 = wmma_f16(aq1, load_B32x16(Ktb, N_, lane, 32, kb + 16), s1);

    // ---- leaky-relu + adjacency mask ----
    float p0[8], p1[8];
#pragma unroll
    for (int r = 0; r < 8; ++r) {
      int rowm = qbase + r + 8 * grp;
      const int* arow = adjb + (size_t)rowm * N_ + kb + ln;
      float v0 = s0[r]; v0 = (v0 > 0.0f) ? v0 : ALPHA_ * v0;
      float v1 = s1[r]; v1 = (v1 > 0.0f) ? v1 : ALPHA_ * v1;
      p0[r] = (arow[0]  > 0) ? v0 : NEGV;
      p1[r] = (arow[16] > 0) ? v1 : NEGV;
    }

    // ---- online softmax (row lives across 16 lanes of the half-group) ----
#pragma unroll
    for (int r = 0; r < 8; ++r) {
      float tm = fmaxf(p0[r], p1[r]);
      tm = fmaxf(tm, __shfl_xor(tm, 1, 32));
      tm = fmaxf(tm, __shfl_xor(tm, 2, 32));
      tm = fmaxf(tm, __shfl_xor(tm, 4, 32));
      tm = fmaxf(tm, __shfl_xor(tm, 8, 32));
      float newm  = fmaxf(mrow[r], tm);
      float scale = __expf(mrow[r] - newm);
      mrow[r] = newm;
      float e0 = __expf(p0[r] - newm);
      float e1 = __expf(p1[r] - newm);
      float rs = e0 + e1;
      rs += __shfl_xor(rs, 1, 32);
      rs += __shfl_xor(rs, 2, 32);
      rs += __shfl_xor(rs, 4, 32);
      rs += __shfl_xor(rs, 8, 32);
      lrow[r] = lrow[r] * scale + rs;
#pragma unroll
      for (int c = 0; c < 4; ++c) acc[c][r] *= scale;
      // scatter P (C layout) into wave-private LDS tile, row-major 16x32
      int rowm = r + 8 * grp;
      ldsP[wave][rowm][ln]      = (_Float16)e0;
      ldsP[wave][rowm][16 + ln] = (_Float16)e1;
    }

    // ---- reload P in A-operand layout (LDS ops are in-order per wave) ----
    {
      int klo = grp << 3;
      const _Float16* pp = &ldsP[wave][ln][klo];
      v8h lo = *(const v8h*)pp;
      v8h hi = *(const v8h*)(pp + 16);
      v16h pA = __builtin_shufflevector(lo, hi,
                  0,1,2,3,4,5,6,7,8,9,10,11,12,13,14,15);
      // ---- acc += P @ V ----
#pragma unroll
      for (int c = 0; c < 4; ++c) {
        v16h bv = load_B32x16(Vb, OUTD, lane, kb, c * 16);
        acc[c] = wmma_f16(pA, bv, acc[c]);
      }
    }
  }

  // ---- epilogue: normalize + concat heads -> [B, N, 512] f32 ----
  float* ob = out + ((size_t)b * N_ + qbase) * HD + (size_t)h * OUTD;
#pragma unroll
  for (int c = 0; c < 4; ++c) {
#pragma unroll
    for (int r = 0; r < 8; ++r) {
      int rowm = r + 8 * grp;
      int d    = c * 16 + ln;
      ob[(size_t)rowm * HD + d] = acc[c][r] / lrow[r];
    }
  }
}

extern "C" void kernel_launch(void* const* d_in, const int* in_sizes, int n_in,
                              void* d_out, int out_size, void* d_ws, size_t ws_size,
                              hipStream_t stream) {
  const float* features = (const float*)d_in[0];   // [4,2048,256] f32
  const int*   adj      = (const int*)  d_in[1];   // [4,2048,2048] i32
  const float* Wq       = (const float*)d_in[2];   // [256,512] f32
  const float* Wk       = (const float*)d_in[3];
  const float* Wv       = (const float*)d_in[4];
  float* out = (float*)d_out;                      // [4,2048,512] f32

  // workspace layout (f16): Xh | Wh[3] | Q | Kt | V  (~30 MB total)
  _Float16* Xh = (_Float16*)d_ws;
  _Float16* Wh = Xh + (size_t)B_ * N_ * IND;             // 2,097,152
  _Float16* Qh = Wh + (size_t)3 * IND * HD;              // + 393,216
  _Float16* Kt = Qh + (size_t)B_ * NH * N_ * OUTD;       // + 4,194,304
  _Float16* Vh = Kt + (size_t)B_ * NH * N_ * OUTD;       // + 4,194,304

  const int nX = B_ * N_ * IND;
  const int nW = IND * HD;
  f32_to_f16_kernel<<<(nX + 255) / 256, 256, 0, stream>>>(features, Xh, nX);
  f32_to_f16_kernel<<<(nW + 255) / 256, 256, 0, stream>>>(Wq, Wh,          nW);
  f32_to_f16_kernel<<<(nW + 255) / 256, 256, 0, stream>>>(Wk, Wh + nW,     nW);
  f32_to_f16_kernel<<<(nW + 255) / 256, 256, 0, stream>>>(Wv, Wh + 2 * nW, nW);

  proj_kernel<<<dim3((B_ * N_) / 16, 3, 1), 256, 0, stream>>>(Xh, Wh, Qh, Kt, Vh);

  attn_kernel<<<dim3(N_ / 128, NH, B_), 256, 0, stream>>>(Qh, Kt, Vh, adj, out);
}
